// OrbitalAttentionPool_22728966930568
// MI455X (gfx1250) — compile-verified, matrix operand order checked by reading, not run on hardware
//
#include <hip/hip_runtime.h>
#include <hip/hip_bf16.h>

typedef __attribute__((ext_vector_type(16))) __bf16 v16bf;
typedef __attribute__((ext_vector_type(8)))  __bf16 v8bf;
typedef __attribute__((ext_vector_type(8)))  float  v8f;

#define HDIM 256
#define HHALF 128
#define WSTRIDE 264   // padded row stride in bf16 elems: 528B = 33*16B, conflict-free

// ---------------- helpers ----------------

static __device__ __forceinline__ v16bf pack16(const float* t) {
  v16bf r;
#pragma unroll
  for (int i = 0; i < 16; ++i) r[i] = (__bf16)t[i];
  return r;
}

// A-fragment (16x32 bf16) built from f32 row-major global memory.
// Lane L (0..15): M=L, holds K = kA..kA+7 (elems 0..7) and kA+16..kA+23 (elems 8..15)
// Lane L (16..31): same rows, kA already shifted by +8 by caller.
static __device__ __forceinline__ v16bf ldA_f32(const float* __restrict__ rowp, int kA) {
  float t[16];
  const float4* p0 = (const float4*)(rowp + kA);
  float4 a = p0[0], b = p0[1];
  const float4* p1 = (const float4*)(rowp + kA + 16);
  float4 c = p1[0], d = p1[1];
  t[0]=a.x; t[1]=a.y; t[2]=a.z; t[3]=a.w; t[4]=b.x; t[5]=b.y; t[6]=b.z; t[7]=b.w;
  t[8]=c.x; t[9]=c.y; t[10]=c.z; t[11]=c.w; t[12]=d.x; t[13]=d.y; t[14]=d.z; t[15]=d.w;
  return pack16(t);
}

// A-fragment from bf16 row-major global memory (two 16B loads).
static __device__ __forceinline__ v16bf ldA_bf16(const __bf16* __restrict__ rowp, int kA) {
  v8bf lo = *(const v8bf*)(rowp + kA);
  v8bf hi = *(const v8bf*)(rowp + kA + 16);
  v16bf r;
#pragma unroll
  for (int i = 0; i < 8; ++i) { r[i] = lo[i]; r[i + 8] = hi[i]; }
  return r;
}

// B-fragment (32x16 bf16) from padded LDS (W^T rows, stride WSTRIDE).
// Address = n*WSTRIDE + kA -> per-ct base VGPR + compile-time DS immediate.
static __device__ __forceinline__ v16bf ldB_lds(const __bf16* sW, int n, int kA) {
  v8bf lo = *(const v8bf*)(sW + n * WSTRIDE + kA);
  v8bf hi = *(const v8bf*)(sW + n * WSTRIDE + kA + 16);
  v16bf r;
#pragma unroll
  for (int i = 0; i < 8; ++i) { r[i] = lo[i]; r[i + 8] = hi[i]; }
  return r;
}

// B-fragment from f32 row-major [K][ld] global weights (lane-coalesced dword loads).
static __device__ __forceinline__ v16bf ldB_g32(const float* __restrict__ W, int ld, int col, int kA) {
  float t[16];
#pragma unroll
  for (int j = 0; j < 8; ++j) t[j]     = W[(size_t)(kA + j) * ld + col];
#pragma unroll
  for (int j = 0; j < 8; ++j) t[8 + j] = W[(size_t)(kA + 16 + j) * ld + col];
  return pack16(t);
}

static __device__ __forceinline__ unsigned encf(float x) {
  unsigned u = __float_as_uint(x);
  return (u & 0x80000000u) ? ~u : (u | 0x80000000u);
}
static __device__ __forceinline__ float decf(unsigned e) {
  unsigned u = (e & 0x80000000u) ? (e ^ 0x80000000u) : ~e;
  return __uint_as_float(u);
}

// ---------------- kernels ----------------

__global__ void init_kernel(unsigned* smaxu, float* denom, float* mol, int M) {
  size_t i = (size_t)blockIdx.x * blockDim.x + threadIdx.x;
  if (i < (size_t)M) { smaxu[i] = 0u; denom[i] = 0.f; }
  if (i < (size_t)M * HDIM) mol[i] = 0.f;
}

// logits = relu(X @ aW1 + ab1) @ aW2 + ab2   (the hot kernel: WMMA bf16)
__global__ __launch_bounds__(256) void attn_logits_kernel(
    const float* __restrict__ X, const float* __restrict__ aW1,
    const float* __restrict__ ab1, const float* __restrict__ aW2,
    const float* __restrict__ ab2, float* __restrict__ logits,
    int N, int numTiles)
{
  extern __shared__ __bf16 sW[];       // aW1^T, bf16, padded rows: 128*264*2 = 67584 B
  // cooperative stage: aW1 is [K=256][n=128] row-major; e = k*128+n (coalesced)
  for (int e = threadIdx.x; e < HDIM * HHALF; e += 256) {
    int k = e >> 7, n = e & 127;
    sW[n * WSTRIDE + k] = (__bf16)aW1[e];
  }

  const int lane = threadIdx.x & 31, wave = threadIdx.x >> 5;
  const bool hi = lane >= 16;
  const int kOff = hi ? 8 : 0;
  const float b2 = ab2[0];

  for (int tile = blockIdx.x; tile < numTiles; tile += gridDim.x) {
    // Barrier every tile: (a) required once after staging, (b) acts as an LDS
    // fence so the compiler cannot hoist the 64 loop-invariant B fragments out
    // of the tile loop (which would spill 512 VGPRs to scratch in the hot loop).
    __syncthreads();

    int rowBase = tile * 128 + wave * 16;
    int rowA = rowBase + (lane & 15);
    if (rowA >= N) rowA = N - 1;
    const float* rowp = X + (size_t)rowA * HDIM;

    v8f z = {0.f,0.f,0.f,0.f,0.f,0.f,0.f,0.f};
    v8f acc[8];
#pragma unroll
    for (int ct = 0; ct < 8; ++ct) acc[ct] = z;

    // software-pipelined: prefetch next A tile + next B fragment
    v16bf a = ldA_f32(rowp, kOff);
#pragma unroll
    for (int ks = 0; ks < 8; ++ks) {
      int kA = ks * 32 + kOff;
      v16bf an = (ks < 7) ? ldA_f32(rowp, kA + 32) : a;
      v16bf b = ldB_lds(sW, (lane & 15), kA);
#pragma unroll
      for (int ct = 0; ct < 8; ++ct) {
        v16bf bn = (ct < 7) ? ldB_lds(sW, (ct + 1) * 16 + (lane & 15), kA) : b;
        acc[ct] = __builtin_amdgcn_wmma_f32_16x16x32_bf16(
            false, a, false, b, (short)0, acc[ct], false, false);
        b = bn;
      }
      a = an;
    }

    // fused: relu(h + ab1) . aW2, then reduce 16 lanes -> rows
    float part[8] = {0.f,0.f,0.f,0.f,0.f,0.f,0.f,0.f};
#pragma unroll
    for (int ct = 0; ct < 8; ++ct) {
      int col = ct * 16 + (lane & 15);
      float bb = ab1[col], ww = aW2[col];
#pragma unroll
      for (int r = 0; r < 8; ++r)
        part[r] += fmaxf(acc[ct][r] + bb, 0.f) * ww;
    }
#pragma unroll
    for (int m = 1; m < 16; m <<= 1)
#pragma unroll
      for (int r = 0; r < 8; ++r) part[r] += __shfl_xor(part[r], m, 32);

    if ((lane & 15) == 0) {
      int rb = rowBase + (hi ? 8 : 0);
#pragma unroll
      for (int r = 0; r < 8; ++r) {
        int row = rb + r;
        if (row < N) logits[row] = part[r] + b2;
      }
    }
  }
}

__global__ void segmax_kernel(const float* __restrict__ logits, const int* __restrict__ batch,
                              unsigned* __restrict__ smaxu, int N) {
  int i = blockIdx.x * blockDim.x + threadIdx.x;
  if (i < N) atomicMax(&smaxu[batch[i]], encf(logits[i]));
}

__global__ void segfin_kernel(const unsigned* __restrict__ smaxu, float* __restrict__ smaxf, int M) {
  int m = blockIdx.x * blockDim.x + threadIdx.x;
  if (m < M) {
    unsigned e = smaxu[m];
    float v = e ? decf(e) : 0.f;
    if (!__builtin_isfinite(v)) v = 0.f;
    smaxf[m] = v;
  }
}

__global__ void exden_kernel(const float* __restrict__ logits, const int* __restrict__ batch,
                             const float* __restrict__ smaxf, float* __restrict__ ex,
                             float* __restrict__ denom, int N) {
  int i = blockIdx.x * blockDim.x + threadIdx.x;
  if (i < N) {
    int b = batch[i];
    float e = __expf(logits[i] - smaxf[b]);
    ex[i] = e;
    atomicAdd(&denom[b], e);
  }
}

__global__ void wnorm_kernel(const float* __restrict__ ex, const int* __restrict__ batch,
                             const float* __restrict__ denom, float* __restrict__ w, int N) {
  int i = blockIdx.x * blockDim.x + threadIdx.x;
  if (i < N) w[i] = ex[i] / denom[batch[i]];
}

// mol[b] += sum_i w[i]*X[i]; sorted batch -> register accumulate, flush at
// segment boundaries with global_atomic_add_f32 (thread t owns column t).
__global__ __launch_bounds__(256) void weighted_kernel(
    const float* __restrict__ X, const int* __restrict__ batch,
    const float* __restrict__ w, float* __restrict__ mol, int N)
{
  int t = threadIdx.x;
  int start = blockIdx.x * 128;
  int end = start + 128; if (end > N) end = N;
  int cnt = end - start;
  __shared__ int   sb[128];
  __shared__ float sw[128];
  if (t < cnt) { sb[t] = batch[start + t]; sw[t] = w[start + t]; }
  __syncthreads();
  float acc = 0.f;
  int cur = sb[0];
  for (int i = 0; i < cnt; ++i) {
    int b = sb[i];
    if (b != cur) {
      atomicAdd(&mol[(size_t)cur * HDIM + t], acc);
      acc = 0.f; cur = b;
    }
    acc += X[(size_t)(start + i) * HDIM + t] * sw[i];
  }
  atomicAdd(&mol[(size_t)cur * HDIM + t], acc);
}

// g1 = relu(mol @ gW1 + gb1), bf16 out   (WMMA, 16 col-tiles)
__global__ __launch_bounds__(256) void mlp1_kernel(
    const float* __restrict__ mol, const float* __restrict__ gW1,
    const float* __restrict__ gb1, __bf16* __restrict__ g1, int M)
{
  const int lane = threadIdx.x & 31, wave = threadIdx.x >> 5;
  const bool hi = lane >= 16;
  int rowTile = blockIdx.x * 128 + wave * 16;
  int rowA = rowTile + (lane & 15);
  if (rowA >= M) rowA = M - 1;
  const float* rowp = mol + (size_t)rowA * HDIM;

  v8f z = {0.f,0.f,0.f,0.f,0.f,0.f,0.f,0.f};
  v8f acc[16];
#pragma unroll
  for (int ct = 0; ct < 16; ++ct) acc[ct] = z;

#pragma unroll
  for (int ks = 0; ks < 8; ++ks) {
    int kA = ks * 32 + (hi ? 8 : 0);
    v16bf a = ldA_f32(rowp, kA);
#pragma unroll
    for (int ct = 0; ct < 16; ++ct) {
      int col = ct * 16 + (lane & 15);
      v16bf b = ldB_g32(gW1, HDIM, col, kA);
      acc[ct] = __builtin_amdgcn_wmma_f32_16x16x32_bf16(
          false, a, false, b, (short)0, acc[ct], false, false);
    }
  }

  int rb = rowTile + (hi ? 8 : 0);
#pragma unroll
  for (int ct = 0; ct < 16; ++ct) {
    int col = ct * 16 + (lane & 15);
    float bb = gb1[col];
#pragma unroll
    for (int r = 0; r < 8; ++r) {
      int row = rb + r;
      if (row < M) g1[(size_t)row * HDIM + col] = (__bf16)fmaxf(acc[ct][r] + bb, 0.f);
    }
  }
}

// out = relu(g1 @ gW2 + gb2) @ gW3 + gb3   (WMMA + fused final dot)
__global__ __launch_bounds__(256) void mlp2_kernel(
    const __bf16* __restrict__ g1, const float* __restrict__ gW2,
    const float* __restrict__ gb2, const float* __restrict__ gW3,
    const float* __restrict__ gb3, float* __restrict__ out, int M)
{
  const int lane = threadIdx.x & 31, wave = threadIdx.x >> 5;
  const bool hi = lane >= 16;
  int rowTile = blockIdx.x * 128 + wave * 16;
  int rowA = rowTile + (lane & 15);
  if (rowA >= M) rowA = M - 1;
  const __bf16* rowp = g1 + (size_t)rowA * HDIM;

  v8f z = {0.f,0.f,0.f,0.f,0.f,0.f,0.f,0.f};
  v8f acc[8];
#pragma unroll
  for (int ct = 0; ct < 8; ++ct) acc[ct] = z;

#pragma unroll
  for (int ks = 0; ks < 8; ++ks) {
    int kA = ks * 32 + (hi ? 8 : 0);
    v16bf a = ldA_bf16(rowp, kA);
#pragma unroll
    for (int ct = 0; ct < 8; ++ct) {
      int col = ct * 16 + (lane & 15);
      v16bf b = ldB_g32(gW2, HHALF, col, kA);
      acc[ct] = __builtin_amdgcn_wmma_f32_16x16x32_bf16(
          false, a, false, b, (short)0, acc[ct], false, false);
    }
  }

  float b3 = gb3[0];
  float part[8] = {0.f,0.f,0.f,0.f,0.f,0.f,0.f,0.f};
#pragma unroll
  for (int ct = 0; ct < 8; ++ct) {
    int col = ct * 16 + (lane & 15);
    float bb = gb2[col], ww = gW3[col];
#pragma unroll
    for (int r = 0; r < 8; ++r)
      part[r] += fmaxf(acc[ct][r] + bb, 0.f) * ww;
  }
#pragma unroll
  for (int m = 1; m < 16; m <<= 1)
#pragma unroll
    for (int r = 0; r < 8; ++r) part[r] += __shfl_xor(part[r], m, 32);

  if ((lane & 15) == 0) {
    int rb = rowTile + (hi ? 8 : 0);
#pragma unroll
    for (int r = 0; r < 8; ++r) {
      int row = rb + r;
      if (row < M) out[row] = part[r] + b3;
    }
  }
}

// ---------------- launch ----------------

extern "C" void kernel_launch(void* const* d_in, const int* in_sizes, int n_in,
                              void* d_out, int out_size, void* d_ws, size_t ws_size,
                              hipStream_t stream)
{
  (void)n_in; (void)ws_size;
  const float* X     = (const float*)d_in[0];
  const int*   batch = (const int*)  d_in[1];
  const float* aW1   = (const float*)d_in[3];
  const float* ab1   = (const float*)d_in[4];
  const float* aW2   = (const float*)d_in[5];
  const float* ab2   = (const float*)d_in[6];
  const float* gW1   = (const float*)d_in[7];
  const float* gb1   = (const float*)d_in[8];
  const float* gW2   = (const float*)d_in[9];
  const float* gb2   = (const float*)d_in[10];
  const float* gW3   = (const float*)d_in[11];
  const float* gb3   = (const float*)d_in[12];
  int N = in_sizes[1];
  int M = out_size;
  float* out = (float*)d_out;

  char* p = (char*)d_ws;
  auto take = [&](size_t bytes) -> char* {
    char* r = p; p += (bytes + 255) & ~(size_t)255; return r;
  };
  float*    logits = (float*)   take((size_t)N * 4);
  float*    ex     = (float*)   take((size_t)N * 4);
  float*    wgt    = (float*)   take((size_t)N * 4);
  unsigned* smaxu  = (unsigned*)take((size_t)M * 4);
  float*    smaxf  = (float*)   take((size_t)M * 4);
  float*    denom  = (float*)   take((size_t)M * 4);
  float*    mol    = (float*)   take((size_t)M * HDIM * 4);
  __bf16*   g1     = (__bf16*)  take((size_t)M * HDIM * 2);

  int numTiles  = (N + 127) / 128;
  int attnGrid  = numTiles < 1024 ? numTiles : 1024;
  int nThreads  = (N + 255) / 256;
  int mThreads  = (M + 255) / 256;
  int molTiles  = (M + 127) / 128;
  int initGrid  = (int)(((size_t)M * HDIM + 255) / 256);
  size_t attnLds = (size_t)HHALF * WSTRIDE * sizeof(__bf16);  // 67584 B < 320KB WGP LDS

  init_kernel<<<initGrid, 256, 0, stream>>>(smaxu, denom, mol, M);
  attn_logits_kernel<<<attnGrid, 256, attnLds, stream>>>(X, aW1, ab1, aW2, ab2, logits, N, numTiles);
  segmax_kernel<<<nThreads, 256, 0, stream>>>(logits, batch, smaxu, N);
  segfin_kernel<<<mThreads, 256, 0, stream>>>(smaxu, smaxf, M);
  exden_kernel<<<nThreads, 256, 0, stream>>>(logits, batch, smaxf, ex, denom, N);
  wnorm_kernel<<<nThreads, 256, 0, stream>>>(ex, batch, denom, wgt, N);
  weighted_kernel<<<numTiles, 256, 0, stream>>>(X, batch, wgt, mol, N);
  mlp1_kernel<<<molTiles, 256, 0, stream>>>(mol, gW1, gb1, g1, M);
  mlp2_kernel<<<molTiles, 256, 0, stream>>>(g1, gW2, gb2, gW3, gb3, out, M);
}
